// RelPartialLearnableMultiHeadAttn_44787918962704
// MI455X (gfx1250) — compile-verified, matrix operand order checked by reading, not run on hardware
//
#include <hip/hip_runtime.h>
#include <hip/hip_bf16.h>

// ---------------------------------------------------------------------------
// Transformer-XL relative multi-head attention for MI455X (gfx1250, wave32).
// - All matmuls: v_wmma_f32_16x16x32_bf16 (fp32 accumulate).
// - Attention fully fused; 16x2048 fp32 score tile resident in LDS.
// - K/RK/V tiles staged with GLOBAL_LOAD_ASYNC_TO_LDS_B128 (ASYNCcnt),
//   double-buffered for copy/compute overlap.
// - prob (512MB, the roofline floor at 23.3 TB/s) streamed with NT stores.
// ---------------------------------------------------------------------------

typedef __bf16 bf16_t;
typedef __attribute__((ext_vector_type(16))) __bf16 v16bf;
typedef __attribute__((ext_vector_type(8)))  __bf16 v8bf;
typedef __attribute__((ext_vector_type(8)))  float  v8f;

#define T_LEN  2048
#define BSZ    2
#define DMODEL 1024
#define NHEAD  16
#define DHEAD  64
#define SCALE  0.125f   // 1/sqrt(64)

// --------------------------- WMMA helpers ----------------------------------
// 16-bit A-matrix 16x32 layout (ISA 7.12.2): lane&15 = row M, lane>>4 selects
// K-halves {kb..kb+7, 16+kb..16+kb+7}, kb = (lane>>4)*8. Each half is one
// contiguous 16B chunk -> load as v8bf (b128).
__device__ inline v16bf load_frag_bf16(const bf16_t* __restrict__ p, int ld) {
  const int lane = threadIdx.x & 31;
  const bf16_t* row = p + (lane & 15) * ld;
  const int kb = (lane >> 4) << 3;
  union { v16bf f; v8bf h[2]; } u;
  u.h[0] = *(const v8bf*)(row + kb);
  u.h[1] = *(const v8bf*)(row + 16 + kb);
  return u.f;
}

__device__ inline v16bf load_frag_f32(const float* __restrict__ p, int ld) {
  const int lane = threadIdx.x & 31;
  const float* row = p + (lane & 15) * ld;
  const int kb = (lane >> 4) << 3;
  v8f a0 = *(const v8f*)(row + kb);
  v8f a1 = *(const v8f*)(row + 16 + kb);
  v16bf f;
#pragma unroll
  for (int i = 0; i < 8; ++i) f[i] = (bf16_t)a0[i];
#pragma unroll
  for (int i = 0; i < 8; ++i) f[8 + i] = (bf16_t)a1[i];
  return f;
}

// K-major B source: element (k,n) at p[k*ld + n]
__device__ inline v16bf load_frag_kmaj(const bf16_t* __restrict__ p, int ld) {
  const int lane = threadIdx.x & 31;
  const int n = lane & 15;
  const int kb = (lane >> 4) << 3;
  v16bf f;
#pragma unroll
  for (int i = 0; i < 8; ++i) f[i] = p[(kb + i) * ld + n];
#pragma unroll
  for (int i = 0; i < 8; ++i) f[8 + i] = p[(16 + kb + i) * ld + n];
  return f;
}

__device__ inline v8f wmma_bf16(v16bf a, v16bf b, v8f c) {
  // (neg_a, A, neg_b, B, c_mod, C, reuse_a, reuse_b)
  return __builtin_amdgcn_wmma_f32_16x16x32_bf16(false, a, false, b,
                                                 (short)0, c, false, false);
}

// ---------------------- CDNA5 async LDS staging ----------------------------
// 16KB tile (128x64 bf16): 1024 b128 transfers, 4 per thread (256 threads).
// VDST VGPR holds the LDS byte address (generic addr low 32 bits == LDS addr).
__device__ inline void async_tile_16k(bf16_t* dst_lds,
                                      const bf16_t* __restrict__ src,
                                      int tid) {
#pragma unroll
  for (int i = 0; i < 4; ++i) {
    const int c = (tid + i * 256) * 8;              // 8 bf16 = 16 bytes
    unsigned lds = (unsigned)(unsigned long long)(dst_lds + c);
    const bf16_t* g = src + c;
    asm volatile("global_load_async_to_lds_b128 %0, %1, off"
                 :: "v"(lds), "v"(g) : "memory");
  }
}
__device__ inline void wait_async0() {
  asm volatile("s_wait_asynccnt 0x0" ::: "memory");
}

// ------------------------- QKV projection ----------------------------------
// heads[m][e] = sum_d w[m][d] * w_qkv[e][d];  m = t*B+b, e in [0,3072)
// Epilogue: split q/k/v, add r_r_bias to q, store bf16 as [b][h][t][dh].
__global__ __launch_bounds__(256) void proj_qkv_kernel(
    const float* __restrict__ w, const float* __restrict__ w_qkv,
    const float* __restrict__ r_r_bias,
    bf16_t* __restrict__ rq_ws, bf16_t* __restrict__ k_ws,
    bf16_t* __restrict__ v_ws) {
  const int wv   = threadIdx.x >> 5;
  const int lane = threadIdx.x & 31;
  const int m0 = blockIdx.x * 16;
  const int n0 = blockIdx.y * 128 + wv * 16;
  v8f acc = {};
  for (int k0 = 0; k0 < DMODEL; k0 += 32) {
    __builtin_prefetch(w + (size_t)m0 * DMODEL + k0 + 64, 0, 0);
    v16bf a  = load_frag_f32(w + (size_t)m0 * DMODEL + k0, DMODEL);
    v16bf bm = load_frag_f32(w_qkv + (size_t)n0 * DMODEL + k0, DMODEL);
    acc = wmma_bf16(a, bm, acc);
  }
  const int e   = n0 + (lane & 15);
  const int sel = e >> 10;               // 0=q, 1=k, 2=v
  const int h   = (e & 1023) >> 6;
  const int dh  = e & 63;
  const int mb  = (lane >> 4) << 3;
  const float bias = (sel == 0) ? r_r_bias[h * DHEAD + dh] : 0.0f;
  bf16_t* dst = (sel == 0) ? rq_ws : ((sel == 1) ? k_ws : v_ws);
#pragma unroll
  for (int i = 0; i < 8; ++i) {
    const int m = m0 + mb + i;
    const int t = m >> 1, b = m & 1;
    dst[((size_t)(b * NHEAD + h) * T_LEN + t) * DHEAD + dh] =
        (bf16_t)(acc[i] + bias);
  }
}

// --------------------------- R projection ----------------------------------
__global__ __launch_bounds__(256) void proj_r_kernel(
    const float* __restrict__ r, const float* __restrict__ w_r,
    bf16_t* __restrict__ rk_ws) {
  const int wv   = threadIdx.x >> 5;
  const int lane = threadIdx.x & 31;
  const int m0 = blockIdx.x * 16;
  const int n0 = blockIdx.y * 128 + wv * 16;
  v8f acc = {};
  for (int k0 = 0; k0 < DMODEL; k0 += 32) {
    v16bf a  = load_frag_f32(r + (size_t)m0 * DMODEL + k0, DMODEL);
    v16bf bm = load_frag_f32(w_r + (size_t)n0 * DMODEL + k0, DMODEL);
    acc = wmma_bf16(a, bm, acc);
  }
  const int e  = n0 + (lane & 15);
  const int h  = e >> 6, dh = e & 63;
  const int mb = (lane >> 4) << 3;
#pragma unroll
  for (int i = 0; i < 8; ++i)
    rk_ws[((size_t)h * T_LEN + (m0 + mb + i)) * DHEAD + dh] = (bf16_t)acc[i];
}

// ------------------------ Fused attention kernel ---------------------------
// One block per (16 q-rows, head, batch). 8 waves / 256 threads.
// LDS (~167KB, legal on CDNA5's 320KB/WGP), double-buffered async staging.
__global__ __launch_bounds__(256) void attn_kernel(
    const bf16_t* __restrict__ rq, const bf16_t* __restrict__ kmat,
    const bf16_t* __restrict__ vmat, const bf16_t* __restrict__ rkmat,
    const unsigned char* __restrict__ mask,  // (T,B) bool
    float* __restrict__ prob_out, bf16_t* __restrict__ av_out) {
  __shared__ float  sc[16 * T_LEN];        // 128 KB score tile
  __shared__ bf16_t kv[2][128 * DHEAD];    //  32 KB K/RK/V double buffer
  __shared__ bf16_t rqs[16 * DHEAD];       //   2 KB
  __shared__ bf16_t pt[16 * 128];          //   4 KB prob bf16 tile
  __shared__ float  red[16 * 16];          //   1 KB

  const int tid = threadIdx.x;
  const int wv = tid >> 5, lane = tid & 31;
  const int nlane = lane & 15, mb = (lane >> 4) << 3;
  const int q0 = blockIdx.x * 16;
  const int h  = blockIdx.y;
  const int b  = blockIdx.z;

  const bf16_t* rq_bh = rq   + (size_t)(b * NHEAD + h) * T_LEN * DHEAD;
  const bf16_t* k_bh  = kmat + (size_t)(b * NHEAD + h) * T_LEN * DHEAD;
  const bf16_t* v_bh  = vmat + (size_t)(b * NHEAD + h) * T_LEN * DHEAD;
  const bf16_t* rk_h  = rkmat + (size_t)h * T_LEN * DHEAD;

  // Stage the 16x64 rq tile (vector copies) + kick off first K tile async.
  for (int i = tid; i < 128; i += 256)
    ((v8bf*)rqs)[i] = ((const v8bf*)(rq_bh + q0 * DHEAD))[i];
  async_tile_16k(kv[0], k_bh, tid);
  __syncthreads();

  // rq A-fragments are invariant across all AC and BD tiles: hoist them.
  const v16bf a0 = load_frag_bf16(rqs + 0, DHEAD);
  const v16bf a1 = load_frag_bf16(rqs + 32, DHEAD);

  // -------- Phase 1: AC = rq . k^T (async double-buffered) ----------------
  for (int kt = 0; kt < 16; ++kt) {
    const int cur = kt & 1;
    wait_async0();
    __syncthreads();
    if (kt + 1 < 16)
      async_tile_16k(kv[cur ^ 1], k_bh + (size_t)(kt + 1) * 128 * DHEAD, tid);
    v8f acc = {};
    acc = wmma_bf16(a0, load_frag_bf16(kv[cur] + wv * 16 * DHEAD, DHEAD), acc);
    acc = wmma_bf16(a1, load_frag_bf16(kv[cur] + wv * 16 * DHEAD + 32, DHEAD), acc);
    const int col = kt * 128 + wv * 16 + nlane;
#pragma unroll
    for (int i = 0; i < 8; ++i) sc[(mb + i) * T_LEN + col] = acc[i];
  }

  // -------- Phase 2: BD = rq . rk^T, rel-shift fused as scatter -----------
  // raw (q,c) -> score[q][c-(T-1-q)] if c >= T-1-q else score[q-1][c+q+1];
  // targets are provably disjoint, so plain LDS RMW is race-free.
  async_tile_16k(kv[0], rk_h, tid);
  for (int rt = 0; rt < 16; ++rt) {
    const int cur = rt & 1;
    wait_async0();
    __syncthreads();
    if (rt + 1 < 16)
      async_tile_16k(kv[cur ^ 1], rk_h + (size_t)(rt + 1) * 128 * DHEAD, tid);
    v8f acc = {};
    acc = wmma_bf16(a0, load_frag_bf16(kv[cur] + wv * 16 * DHEAD, DHEAD), acc);
    acc = wmma_bf16(a1, load_frag_bf16(kv[cur] + wv * 16 * DHEAD + 32, DHEAD), acc);
    const int c = rt * 128 + wv * 16 + nlane;
#pragma unroll
    for (int i = 0; i < 8; ++i) {
      const int ml = mb + i;
      const int q = q0 + ml;
      const int thr = (T_LEN - 1) - q;
      if (c >= thr)      sc[ml * T_LEN + (c - thr)]         += acc[i];
      else if (ml > 0)   sc[(ml - 1) * T_LEN + (c + q + 1)] += acc[i];
      // ml==0 contribution belongs to the previous block's row-15 strip.
    }
    __syncthreads();
  }

  // Cross-tile strip: BD row q0+16 contributes to local score row 15.
  {
    const int qp = q0 + 16;
    if (qp < T_LEN) {
      const int cmax = T_LEN - 2 - qp;
      const bf16_t* arow = rq_bh + (size_t)qp * DHEAD;
      for (int c = tid; c <= cmax; c += 256) {
        const bf16_t* brow = rk_h + (size_t)c * DHEAD;
        float s = 0.f;
#pragma unroll
        for (int d = 0; d < DHEAD; ++d) s += (float)arow[d] * (float)brow[d];
        sc[15 * T_LEN + (c + qp + 1)] += s;
      }
    }
  }
  __syncthreads();

  // -------- Phase 3: scale, mask, softmax; stream prob (NT stores) --------
  const int row = tid >> 4, ln = tid & 15;
  float* srow = sc + row * T_LEN;
  float mx = -INFINITY;
  for (int k = ln; k < T_LEN; k += 16) {
    float v = srow[k] * SCALE;
    if (mask[k * BSZ + b]) v = -INFINITY;
    srow[k] = v;
    mx = fmaxf(mx, v);
  }
  red[row * 16 + ln] = mx;
  __syncthreads();
#pragma unroll
  for (int j = 0; j < 16; ++j) mx = fmaxf(mx, red[row * 16 + j]);
  __syncthreads();
  float sum = 0.f;
  for (int k = ln; k < T_LEN; k += 16) {
    const float e = __expf(srow[k] - mx);
    srow[k] = e;
    sum += e;
  }
  red[row * 16 + ln] = sum;
  __syncthreads();
  float tot = 0.f;
#pragma unroll
  for (int j = 0; j < 16; ++j) tot += red[row * 16 + j];
  const float inv = 1.0f / tot;
  float* prow = prob_out +
      ((size_t)(b * NHEAD + h) * T_LEN + (q0 + row)) * T_LEN;
  for (int k = ln; k < T_LEN; k += 16) {
    const float p = srow[k] * inv;
    srow[k] = p;
    __builtin_nontemporal_store(p, prow + k);   // don't pollute L2
  }
  __syncthreads();

  // -------- Phase 4: attn_vec = prob . v (async double-buffered) ----------
  v8f oacc = {};
  async_tile_16k(kv[0], v_bh, tid);
  for (int kt = 0; kt < 16; ++kt) {
    const int cur = kt & 1;
    wait_async0();
    __syncthreads();
    if (kt + 1 < 16)
      async_tile_16k(kv[cur ^ 1], v_bh + (size_t)(kt + 1) * 128 * DHEAD, tid);
    // convert prob chunk to bf16 tile (vectorized: 8 contiguous per thread)
    {
      const int r4 = tid >> 4;
      const int c4 = (tid & 15) * 8;
      const float* s = sc + r4 * T_LEN + kt * 128 + c4;
      v8bf o;
#pragma unroll
      for (int i = 0; i < 8; ++i) o[i] = (bf16_t)s[i];
      *(v8bf*)(pt + r4 * 128 + c4) = o;
    }
    __syncthreads();
    if (wv < 4) {                          // 4 waves cover the 64 dh columns
#pragma unroll
      for (int ks = 0; ks < 128; ks += 32) {
        v16bf a  = load_frag_bf16(pt + ks, 128);
        v16bf bm = load_frag_kmaj(kv[cur] + ks * DHEAD + wv * 16, DHEAD);
        oacc = wmma_bf16(a, bm, oacc);
      }
    }
  }
  if (wv < 4) {
    const int dh = wv * 16 + nlane;
#pragma unroll
    for (int i = 0; i < 8; ++i) {
      const int q = q0 + mb + i;
      av_out[((size_t)q * BSZ + b) * (NHEAD * DHEAD) + h * DHEAD + dh] =
          (bf16_t)oacc[i];
    }
  }
}

// ------------------------- Output projection -------------------------------
// out[m][d] = sum_e av[m][e] * w_o[d][e]
__global__ __launch_bounds__(256) void outproj_kernel(
    const bf16_t* __restrict__ av, const float* __restrict__ w_o,
    float* __restrict__ out) {
  const int wv   = threadIdx.x >> 5;
  const int lane = threadIdx.x & 31;
  const int m0 = blockIdx.x * 16;
  const int n0 = blockIdx.y * 128 + wv * 16;
  v8f acc = {};
  for (int k0 = 0; k0 < DMODEL; k0 += 32) {
    v16bf a  = load_frag_bf16(av + (size_t)m0 * DMODEL + k0, DMODEL);
    v16bf bm = load_frag_f32(w_o + (size_t)n0 * DMODEL + k0, DMODEL);
    acc = wmma_bf16(a, bm, acc);
  }
  const int n  = n0 + (lane & 15);
  const int mb = (lane >> 4) << 3;
#pragma unroll
  for (int i = 0; i < 8; ++i)
    out[(size_t)(m0 + mb + i) * DMODEL + n] = acc[i];
}

// ------------------------------ launcher -----------------------------------
extern "C" void kernel_launch(void* const* d_in, const int* in_sizes, int n_in,
                              void* d_out, int out_size, void* d_ws,
                              size_t ws_size, hipStream_t stream) {
  (void)in_sizes; (void)n_in; (void)out_size; (void)ws_size;
  const float* w        = (const float*)d_in[0];   // (T,B,D)
  const float* r        = (const float*)d_in[1];   // (T,D)
  const float* w_qkv    = (const float*)d_in[2];   // (3*H*DH, D)
  const float* w_r      = (const float*)d_in[3];   // (H*DH, D)
  const float* w_o      = (const float*)d_in[4];   // (D, H*DH)
  const float* r_r_bias = (const float*)d_in[5];   // (H, DH)
  const unsigned char* attn_mask = (const unsigned char*)d_in[6];  // (T,B) bool

  float* out  = (float*)d_out;                               // (T,B,D)
  float* prob = out + (size_t)T_LEN * BSZ * DMODEL;          // (B,H,T,T)

  // Workspace layout (36 MB total, bf16 intermediates):
  char* wsb = (char*)d_ws;
  bf16_t* rq_ws = (bf16_t*)(wsb);                        // [B][H][T][DH] 8MB
  bf16_t* k_ws  = (bf16_t*)(wsb + (size_t)8  * 1048576); // [B][H][T][DH] 8MB
  bf16_t* v_ws  = (bf16_t*)(wsb + (size_t)16 * 1048576); // [B][H][T][DH] 8MB
  bf16_t* rk_ws = (bf16_t*)(wsb + (size_t)24 * 1048576); // [H][T][DH]    4MB
  bf16_t* av_ws = (bf16_t*)(wsb + (size_t)28 * 1048576); // [T][B][H*DH]  8MB

  // 1) QKV projection (M=4096, N=3072, K=1024)
  proj_qkv_kernel<<<dim3(256, 24), 256, 0, stream>>>(
      w, w_qkv, r_r_bias, rq_ws, k_ws, v_ws);
  // 2) R projection (M=2048, N=1024, K=1024)
  proj_r_kernel<<<dim3(128, 8), 256, 0, stream>>>(r, w_r, rk_ws);
  // 3) Fused attention: one block per (16 q-rows, head, batch)
  attn_kernel<<<dim3(T_LEN / 16, NHEAD, BSZ), 256, 0, stream>>>(
      rq_ws, k_ws, v_ws, rk_ws, attn_mask, prob, av_ws);
  // 4) Output projection (M=4096, N=1024, K=1024)
  outproj_kernel<<<dim3(256, 8), 256, 0, stream>>>(av_ws, w_o, out);
}
